// LSTM_11819749998795
// MI455X (gfx1250) — matrix operand order repaired and checked by reading, not compiled, not run on hardware
//
#include <hip/hip_runtime.h>
#include <cstddef>

// ---------------------------------------------------------------------------
// LSTM on MI455X (gfx1250): bf16 WMMA everywhere.
//  Phase 1: xproj = x @ [Wxi|Wxf|Wxg|Wxo]  (one big 32768x4096x1024 GEMM)
//  Phase 2: persistent kernel, 32 WGs, Wh^T slice resident in LDS (258KB/WG),
//           cell state in WMMA-C-layout VGPRs, grid barrier per time step.
// ---------------------------------------------------------------------------

typedef __attribute__((ext_vector_type(16))) __bf16 v16bf;
typedef __attribute__((ext_vector_type(8)))  float  v8f;

#define B_   64
#define T_   512
#define H_   1024
#define G4_  4096
#define LP_  1032   // LDS row pitch in bf16 elems (pad 16B -> bank stride 4)

// ---- bf16 helpers ---------------------------------------------------------
static __device__ __forceinline__ unsigned short f2bfu(float f) {
  unsigned u; __builtin_memcpy(&u, &f, 4);
  u = (u + 0x7FFFu + ((u >> 16) & 1u)) >> 16;
  return (unsigned short)u;
}
static __device__ __forceinline__ float bfu2f(unsigned short h) {
  unsigned u = ((unsigned)h) << 16;
  float f; __builtin_memcpy(&f, &u, 4);
  return f;
}
static __device__ __forceinline__ v16bf mkfrag(uint4 lo, uint4 hi) {
  v16bf v;
  __builtin_memcpy(&v, &lo, 16);
  __builtin_memcpy(reinterpret_cast<char*>(&v) + 16, &hi, 16);
  return v;
}
// A fragment (16x32 bf16, row-major src): lane = row M, 16 contiguous K at half*16
static __device__ __forceinline__ v16bf ldfragA(const unsigned short* p, int hf) {
  uint4 lo = *reinterpret_cast<const uint4*>(p + hf * 16);
  uint4 hi = *reinterpret_cast<const uint4*>(p + hf * 16 + 8);
  return mkfrag(lo, hi);
}
// B fragment (32x16 bf16 from W^T [N][K]): lane = col N, 16 contiguous K at half*16
static __device__ __forceinline__ v16bf ldfragB(const unsigned short* p, int hf) {
  const uint4* q = reinterpret_cast<const uint4*>(p + hf * 16);
  return mkfrag(q[0], q[1]);
}
static __device__ __forceinline__ v8f wmma_bf16(v16bf a, v16bf b, v8f c) {
  return __builtin_amdgcn_wmma_f32_16x16x32_bf16(false, a, false, b, (short)0, c,
                                                 false, false);
}
static __device__ __forceinline__ float sigm(float x) {
  return 1.f / (1.f + __expf(-x));
}
static __device__ __forceinline__ float tanh_fast(float x) {
  float e = __expf(2.f * x);
  return 1.f - 2.f / (e + 1.f);
}

// ---- K0: zero h ping-pong buffers + grid barrier counter ------------------
__global__ void k_init(unsigned long long* hz, unsigned* cnt) {
  int idx = blockIdx.x * 256 + threadIdx.x;      // 128 blocks * 256 thr * 8B
  hz[idx] = 0ull;                                // 2 * 64*1024 bf16 = 262144 B
  if (idx == 0) *cnt = 0u;
}

// ---- K1: transpose-convert 8 weight matrices [1024x1024] f32 -> W^T bf16 --
__global__ void k_wtrans(const float* s0, const float* s1, const float* s2,
                         const float* s3, const float* s4, const float* s5,
                         const float* s6, const float* s7,
                         unsigned short* wxt, unsigned short* wht) {
  __shared__ float tile[32][33];
  const float* srcs[8] = {s0, s1, s2, s3, s4, s5, s6, s7};
  int z = blockIdx.z;
  const float* S = srcs[z];
  unsigned short* D = ((z < 4) ? wxt : wht) + (size_t)(z & 3) * H_ * H_;
  int tx = threadIdx.x, ty = threadIdx.y;
  int kb = blockIdx.x * 32, nb = blockIdx.y * 32;
#pragma unroll
  for (int i = 0; i < 4; ++i)
    tile[ty + i * 8][tx] = S[(size_t)(kb + ty + i * 8) * H_ + nb + tx];
  __syncthreads();
#pragma unroll
  for (int i = 0; i < 4; ++i)
    D[(size_t)(nb + ty + i * 8) * H_ + kb + tx] = f2bfu(tile[tx][ty + i * 8]);
}

// ---- K2: x f32 -> bf16 ----------------------------------------------------
__global__ void k_xcvt(const float* __restrict__ x, unsigned short* __restrict__ xb) {
  size_t idx = (size_t)blockIdx.x * 256 + threadIdx.x;   // one uint4 (8 bf16) each
  const float4* src = reinterpret_cast<const float4*>(x) + idx * 2;
  float4 a = src[0], b = src[1];
  unsigned short o[8] = {f2bfu(a.x), f2bfu(a.y), f2bfu(a.z), f2bfu(a.w),
                         f2bfu(b.x), f2bfu(b.y), f2bfu(b.z), f2bfu(b.w)};
  uint4 pk; __builtin_memcpy(&pk, o, 16);
  reinterpret_cast<uint4*>(xb)[idx] = pk;
}

// ---- K3: xproj GEMM  [32768 x 1024] x [1024 x 4096] -> bf16 ---------------
// WG tile 64(M) x 128(N); 8 waves: wave = (M-tile, half of N) -> 4 accumulators
__global__ void __launch_bounds__(256)
k_xproj(const unsigned short* __restrict__ xb, const unsigned short* __restrict__ wxt,
        unsigned short* __restrict__ xp) {
  int tid = threadIdx.x, wave = tid >> 5, lane = tid & 31;
  int hf = lane >> 4, col = lane & 15;
  int mb = blockIdx.x * 64 + (wave >> 1) * 16;
  int nb = blockIdx.y * 128 + (wave & 1) * 64;

  v8f acc[4] = {};
  const unsigned short* arow = xb + (size_t)(mb + col) * H_;
#pragma unroll 4
  for (int kc = 0; kc < 32; ++kc) {
    int kb = kc * 32;
    v16bf a = ldfragA(arow + kb, hf);
#pragma unroll
    for (int u = 0; u < 4; ++u) {
      const unsigned short* bp = wxt + (size_t)(nb + u * 16 + col) * H_ + kb;
      acc[u] = wmma_bf16(a, ldfragB(bp, hf), acc[u]);
    }
  }
#pragma unroll
  for (int u = 0; u < 4; ++u)
#pragma unroll
    for (int v = 0; v < 8; ++v) {
      int row = mb + hf * 8 + v;
      int cg  = nb + u * 16 + col;
      xp[(size_t)row * G4_ + cg] = f2bfu(acc[u][v]);
    }
}

// ---- K4: persistent recurrent kernel --------------------------------------
// 32 WGs; WG g owns hidden cols [g*32, g*32+32). LDS holds Wh^T rows for all
// 4 gates of that slice: 128 rows x 1024 K bf16, pitch 1032 (258 KB).
// Wave w: mtile = w>>1 (16 batch rows), jt = w&1 (16 hidden cols); its 4
// accumulators are i/f/g/o for the SAME (row,col) set -> pure per-lane update.
__global__ void __launch_bounds__(256)
k_lstm(const unsigned short* __restrict__ xp, const unsigned short* __restrict__ wht,
       const float* __restrict__ bi, const float* __restrict__ pbf,
       const float* __restrict__ bg, const float* __restrict__ bo,
       unsigned short* __restrict__ hbuf, unsigned* __restrict__ cnt,
       float* __restrict__ hseq, float* __restrict__ ht_out,
       float* __restrict__ ct_out) {
  extern __shared__ unsigned short lds[];
  const int tid = threadIdx.x, wave = tid >> 5, lane = tid & 31;
  const int hf = lane >> 4, col = lane & 15;
  const int g = blockIdx.x, jbase = g * 32;
  const int mbase = (wave >> 1) * 16, jt = wave & 1;
  const int nch = jbase + jt * 16 + col;   // hidden column this lane owns

  // Stage Wh^T slice into LDS (once; resident for all 512 steps).
  for (int idx = tid; idx < 128 * 128; idx += 256) {
    int r  = idx >> 7;            // 0..127 = gate*32 + j
    int c8 = (idx & 127) * 8;     // 8 bf16 chunk
    int n  = (r >> 5) * H_ + jbase + (r & 31);
    uint4 v = *reinterpret_cast<const uint4*>(wht + (size_t)n * H_ + c8);
    *reinterpret_cast<uint4*>(&lds[r * LP_ + c8]) = v;
  }
  __syncthreads();

  const float bias[4] = {bi[nch], pbf[nch], bg[nch], bo[nch]};
  v8f cst = {};   // cell state, WMMA C layout

  for (int t = 0; t < T_; ++t) {
    const unsigned short* hprev = hbuf + (size_t)(t & 1) * (B_ * H_);
    unsigned short*       hnext = hbuf + (size_t)((t + 1) & 1) * (B_ * H_);

    v8f acc[4] = {};
    const unsigned short* arow = hprev + (size_t)(mbase + col) * H_;
#pragma unroll 4
    for (int kc = 0; kc < 32; ++kc) {
      int kb = kc * 32;
      v16bf a = ldfragA(arow + kb, hf);
#pragma unroll
      for (int gate = 0; gate < 4; ++gate) {
        int boff = (gate * 32 + jt * 16 + col) * LP_ + kb + hf * 16;
        uint4 lo = *reinterpret_cast<const uint4*>(&lds[boff]);
        uint4 hi = *reinterpret_cast<const uint4*>(&lds[boff + 8]);
        acc[gate] = wmma_bf16(a, mkfrag(lo, hi), acc[gate]);
      }
    }

#pragma unroll
    for (int v = 0; v < 8; ++v) {
      int b = mbase + hf * 8 + v;
      size_t xrow = ((size_t)b * T_ + t) * G4_;
      float gi = sigm(acc[0][v] + bfu2f(xp[xrow + 0 * H_ + nch]) + bias[0]);
      float gf = sigm(acc[1][v] + bfu2f(xp[xrow + 1 * H_ + nch]) + bias[1]);
      float gg = tanh_fast(acc[2][v] + bfu2f(xp[xrow + 2 * H_ + nch]) + bias[2]);
      float go = sigm(acc[3][v] + bfu2f(xp[xrow + 3 * H_ + nch]) + bias[3]);
      float cn = gf * cst[v] + gi * gg;
      cst[v] = cn;
      float hn = go * tanh_fast(cn);
      hnext[(size_t)b * H_ + nch] = f2bfu(hn);
      hseq[((size_t)b * T_ + t) * H_ + nch] = hn;
      if (t == T_ - 1) {
        ht_out[(size_t)b * H_ + nch] = hn;
        ct_out[(size_t)b * H_ + nch] = cn;
      }
    }

    // Grid-wide step barrier: release stores, arrive, acquire-spin.
    __threadfence();
    __syncthreads();
    if (tid == 0) {
      __hip_atomic_fetch_add(cnt, 1u, __ATOMIC_RELEASE, __HIP_MEMORY_SCOPE_AGENT);
      unsigned target = 32u * (unsigned)(t + 1);
      while (__hip_atomic_load(cnt, __ATOMIC_ACQUIRE, __HIP_MEMORY_SCOPE_AGENT) <
             target)
        __builtin_amdgcn_s_sleep(1);
    }
    __syncthreads();
  }
}

// ---------------------------------------------------------------------------
extern "C" void kernel_launch(void* const* d_in, const int* in_sizes, int n_in,
                              void* d_out, int out_size, void* d_ws, size_t ws_size,
                              hipStream_t stream) {
  (void)in_sizes; (void)n_in; (void)out_size; (void)ws_size;
  const float* x   = (const float*)d_in[0];
  const float* wii = (const float*)d_in[1];
  const float* whi = (const float*)d_in[2];
  const float* bi  = (const float*)d_in[3];
  const float* wif = (const float*)d_in[4];
  const float* whf = (const float*)d_in[5];
  const float* pbf = (const float*)d_in[6];
  const float* wig = (const float*)d_in[7];
  const float* whg = (const float*)d_in[8];
  const float* bg  = (const float*)d_in[9];
  const float* wio = (const float*)d_in[10];
  const float* who = (const float*)d_in[11];
  const float* bo  = (const float*)d_in[12];

  float* out   = (float*)d_out;
  float* hseq  = out;                                  // [64,512,1024]
  float* ht_o  = out + (size_t)B_ * T_ * H_;           // [64,1024]
  float* ct_o  = ht_o + (size_t)B_ * H_;               // [64,1024]

  // workspace layout (bytes)
  char* ws = (char*)d_ws;
  unsigned*       cnt = (unsigned*)ws;                               // 256 B
  unsigned short* hb  = (unsigned short*)(ws + 256);                 // 262144 B
  unsigned short* wxt = (unsigned short*)(ws + 262400);              // 8 MB
  unsigned short* wht = (unsigned short*)(ws + 262400 + 8388608);    // 8 MB
  unsigned short* xb  = (unsigned short*)(ws + 262400 + 2 * 8388608);          // 64 MB
  unsigned short* xpj = (unsigned short*)(ws + 262400 + 2 * 8388608 + 67108864); // 256 MB

  const int smem = 128 * LP_ * 2;   // 264192 B dynamic LDS for k_lstm
  hipFuncSetAttribute((const void*)k_lstm,
                      hipFuncAttributeMaxDynamicSharedMemorySize, smem);

  k_init<<<128, 256, 0, stream>>>((unsigned long long*)hb, cnt);
  k_wtrans<<<dim3(32, 32, 8), dim3(32, 8), 0, stream>>>(
      wii, wif, wig, wio, whi, whf, whg, who, wxt, wht);
  k_xcvt<<<(B_ * T_ * H_) / (256 * 8), 256, 0, stream>>>(x, xb);
  k_xproj<<<dim3((B_ * T_) / 64, G4_ / 128), 256, 0, stream>>>(xb, wxt, xpj);
  k_lstm<<<32, 256, smem, stream>>>(xpj, wht, bi, pbf, bg, bo, hb, cnt,
                                    hseq, ht_o, ct_o);
}
